// UNetSphericalHealpixDeep_67869073211455
// MI455X (gfx1250) — matrix operand order repaired and checked by reading, not compiled, not run on hardware
//
#include <hip/hip_runtime.h>
#include <hip/hip_bf16.h>

// ---------------------------------------------------------------------------
// CDNA5 (gfx1250) spherical Chebyshev U-Net forward.
// cheb_conv == GEMM  C[M,N] = A[M,Kd] @ W[Kd,N],  M=B*V (row b*V+v), Kd=4*Ci.
// GEMM: v_wmma_f32_16x16x32_bf16, one wave32 per 16x(16*NT) output strip,
// with an explicit 2-deep software pipeline (ping-pong fragment buffers) so
// K-step s+1 loads overlap K-step s WMMAs (staggered s_wait_loadcnt).
// ---------------------------------------------------------------------------

typedef __attribute__((ext_vector_type(16))) __bf16 v16bf;
typedef __attribute__((ext_vector_type(8)))  float  v8f;

#define BN_EPS 1e-5f

// ---------------- weight pack: W[4,Ci,Co] (f32) -> Wt[Npad,4*Ci] (bf16, B^T)
__global__ void pack_w_kernel(const float* __restrict__ W, __bf16* __restrict__ Wt,
                              int Ci, int Co, int Npad) {
  int Kd = 4 * Ci;
  int i = blockIdx.x * blockDim.x + threadIdx.x;
  if (i >= Npad * Kd) return;
  int o = i / Kd, kd = i % Kd;
  int k = kd / Ci, ci = kd % Ci;
  float w = (o < Co) ? W[((size_t)k * Ci + ci) * Co + o] : 0.0f;
  Wt[i] = (__bf16)w;
}

// ---------------- Chebyshev basis, order 0: A[m, 0:Ci] = x (row m = b*V+v)
__global__ void cheb0_kernel(const float* __restrict__ x, __bf16* __restrict__ A,
                             int M, int Ci, int Kd) {
  int i = blockIdx.x * blockDim.x + threadIdx.x;
  if (i >= M * Ci) return;
  int m = i / Ci, ci = i % Ci;
  A[(size_t)m * Kd + ci] = (__bf16)x[i];
}

// ---------------- Chebyshev recursion: block k from blocks k-1 (gather) & k-2
__global__ void cheb_lap_kernel(__bf16* __restrict__ A, const int* __restrict__ idx,
                                const float* __restrict__ val,
                                int M, int V, int Ci, int Kd, int k) {
  int i = blockIdx.x * blockDim.x + threadIdx.x;
  if (i >= M * Ci) return;
  int m = i / Ci, ci = i % Ci;
  int v = m % V, base = m - v;         // base = b*V
  float s = 0.0f;
#pragma unroll
  for (int j = 0; j < 8; ++j) {
    int nv = idx[v * 8 + j];
    s += val[v * 8 + j] * (float)A[(size_t)(base + nv) * Kd + (k - 1) * Ci + ci];
  }
  float r = (k == 1) ? s
                     : 2.0f * s - (float)A[(size_t)m * Kd + (k - 2) * Ci + ci];
  A[(size_t)m * Kd + k * Ci + ci] = (__bf16)r;
}

// ---------------- WMMA GEMM: one wave32 per 16 x (16*NT) output strip.
// A: [M,Kd] bf16 row-major.  Wt: [Npad,Kd] bf16 (B transposed).
// Y: [M,Nout] f32 (Nout may be < Npad for the padded final layer).
template <int NT>
__global__ void __launch_bounds__(32)
wmma_gemm_kernel(const __bf16* __restrict__ A, const __bf16* __restrict__ Wt,
                 float* __restrict__ Y, const float* __restrict__ bias,
                 int M, int Kd, int Npad, int Nout) {
  const int mt  = blockIdx.x;
  const int nt0 = blockIdx.y * NT;
  const int lane = threadIdx.x;
  const int row  = lane & 15;    // A: M-row / B,C: N-col
  const int half = lane >> 4;

  // ISA 7.12.2 layouts:
  //  A 16x32 bf16: lane holds row (lane&15); dwords 0-3: K=half*8..+7,
  //                dwords 4-7: K=16+half*8..+7  -> two contiguous 16B runs.
  //  B 32x16 bf16 (Wt rows): lane holds col (lane&15);
  //                dwords 0-3: K=half*8..+7, dwords 4-7: K=16+half*8..+7.
  const __bf16* arow = A + (size_t)(mt * 16 + row) * Kd + half * 8;
  const __bf16* brow[NT];
#pragma unroll
  for (int t = 0; t < NT; ++t)
    brow[t] = Wt + (size_t)((nt0 + t) * 16 + row) * Kd + half * 8;

  v8f acc[NT];
#pragma unroll
  for (int t = 0; t < NT; ++t) acc[t] = {};

  union Frag { uint4 u[2]; v16bf v; };

  auto load_step = [&](int k0, Frag& a, Frag (&b)[NT]) {
    a.u[0] = *(const uint4*)(arow + k0);
    a.u[1] = *(const uint4*)(arow + k0 + 16);
#pragma unroll
    for (int t = 0; t < NT; ++t) {
      b[t].u[0] = *(const uint4*)(brow[t] + k0);
      b[t].u[1] = *(const uint4*)(brow[t] + k0 + 16);
    }
  };
  auto mma_step = [&](Frag& a, Frag (&b)[NT]) {
#pragma unroll
    for (int t = 0; t < NT; ++t) {
      acc[t] = __builtin_amdgcn_wmma_f32_16x16x32_bf16(
          /*neg_a=*/false, a.v, /*neg_b=*/false, b[t].v,
          /*c_mod=*/(short)0, acc[t], /*reuse_a=*/false, /*reuse_b=*/false);
    }
  };

  // 2-deep software pipeline over K-steps of 32 (ping-pong A/B buffers).
  const int nsteps = Kd >> 5;
  Frag aA, bA[NT], aB, bB[NT];
  load_step(0, aA, bA);
  int s = 0;
  for (; s + 2 <= nsteps; s += 2) {
    load_step((s + 1) << 5, aB, bB);      // prefetch step s+1
    mma_step(aA, bA);                     // compute step s
    if (s + 2 < nsteps)
      load_step((s + 2) << 5, aA, bA);    // prefetch step s+2
    mma_step(aB, bB);                     // compute step s+1
  }
  if (s < nsteps)                         // odd tail (conv11: nsteps == 1)
    mma_step(aA, bA);

  // C layout: VGPR r -> row half*8 + r, col = lane&15.
#pragma unroll
  for (int t = 0; t < NT; ++t) {
    union { v8f v; float f[8]; } cu;
    cu.v = acc[t];
    const int col = (nt0 + t) * 16 + row;
    if (col < Nout) {
      float bv = bias ? bias[col] : 0.0f;
#pragma unroll
      for (int r = 0; r < 8; ++r)
        Y[(size_t)(mt * 16 + half * 8 + r) * Nout + col] = cu.f[r] + bv;
    }
  }
}

// ---------------- BN stats: one block per channel -> stats[2*ch]={sum,sumsq}
__global__ void stats_kernel(const float* __restrict__ Y, float* __restrict__ stats,
                             int M, int N) {
  __shared__ float ss[256], sq[256];
  int ch = blockIdx.x, tid = threadIdx.x;
  float s = 0.0f, q = 0.0f;
  for (int m = tid; m < M; m += 256) {
    float y = Y[(size_t)m * N + ch];
    s += y; q += y * y;
  }
  ss[tid] = s; sq[tid] = q;
  __syncthreads();
  for (int o = 128; o > 0; o >>= 1) {
    if (tid < o) { ss[tid] += ss[tid + o]; sq[tid] += sq[tid + o]; }
    __syncthreads();
  }
  if (tid == 0) { stats[2 * ch] = ss[0]; stats[2 * ch + 1] = sq[0]; }
}

// ---------------- BN (training-mode stats) + ReLU
__global__ void bn_relu_kernel(const float* __restrict__ Y, const float* __restrict__ stats,
                               const float* __restrict__ g, const float* __restrict__ be,
                               float* __restrict__ out, int M, int N) {
  int i = blockIdx.x * blockDim.x + threadIdx.x;
  if (i >= M * N) return;
  int ch = i % N;
  float inv = 1.0f / (float)M;
  float mu  = stats[2 * ch] * inv;
  float var = stats[2 * ch + 1] * inv - mu * mu;
  float y = (Y[i] - mu) * rsqrtf(var + BN_EPS) * g[ch] + be[ch];
  out[i] = fmaxf(y, 0.0f);
}

// ---------------- healpix max-pool (kernel 4, per channel) with argmax
__global__ void pool_kernel(const float* __restrict__ x, float* __restrict__ vals,
                            int* __restrict__ idxo, int B, int V, int C) {
  int P = V / 4;
  int i = blockIdx.x * blockDim.x + threadIdx.x;
  if (i >= B * P * C) return;
  int c = i % C, bp = i / C;
  int p = bp % P, b = bp / P;
  const float* xp = x + ((size_t)(b * V + p * 4)) * C + c;
  float best = xp[0]; int bi = 0;
#pragma unroll
  for (int j = 1; j < 4; ++j) {
    float t = xp[(size_t)j * C];
    if (t > best) { best = t; bi = j; }   // first-max semantics (argmax)
  }
  vals[i] = best;
  idxo[i] = bi;
}

// ---------------- max-unpool (scatter to argmax slot) + channel concat
__global__ void unpool_concat_kernel(const float* __restrict__ d, const int* __restrict__ idx,
                                     const float* __restrict__ e, float* __restrict__ out,
                                     int B, int P, int C1, int C2) {
  int V = P * 4, C = C1 + C2;
  int i = blockIdx.x * blockDim.x + threadIdx.x;
  if (i >= B * V * C) return;
  int c = i % C, bv = i / C;
  int v = bv % V, b = bv / V;
  float r;
  if (c < C1) {
    int p = v >> 2, q = v & 3;
    size_t di = ((size_t)(b * P + p)) * C1 + c;
    r = (idx[di] == q) ? d[di] : 0.0f;
  } else {
    r = e[((size_t)(b * V + v)) * C2 + (c - C1)];
  }
  out[i] = r;
}

// ===========================================================================
// Host orchestration
// ===========================================================================
static inline int cdiv(int a, int b) { return (a + b - 1) / b; }

static void cheb_gemm(hipStream_t stream, const float* xin, int M, int V,
                      int Ci, int Co, int Npad, int Nout,
                      const float* w, const float* bias,
                      const int* idx, const float* val,
                      __bf16* A, __bf16* Wt, float* Y) {
  int Kd = 4 * Ci;
  int nW = Npad * Kd;
  pack_w_kernel<<<cdiv(nW, 256), 256, 0, stream>>>(w, Wt, Ci, Co, Npad);
  int nE = M * Ci;
  cheb0_kernel<<<cdiv(nE, 256), 256, 0, stream>>>(xin, A, M, Ci, Kd);
  for (int k = 1; k < 4; ++k)
    cheb_lap_kernel<<<cdiv(nE, 256), 256, 0, stream>>>(A, idx, val, M, V, Ci, Kd, k);

  int ntiles = Npad / 16;
  if (ntiles % 4 == 0) {
    dim3 grid(M / 16, ntiles / 4);
    wmma_gemm_kernel<4><<<grid, 32, 0, stream>>>(A, Wt, Y, bias, M, Kd, Npad, Nout);
  } else {
    dim3 grid(M / 16, ntiles);
    wmma_gemm_kernel<1><<<grid, 32, 0, stream>>>(A, Wt, Y, bias, M, Kd, Npad, Nout);
  }
}

static void conv_block(hipStream_t stream, const float* xin, int M, int V,
                       int Ci, int Co,
                       const float* w, const float* g, const float* be,
                       const int* idx, const float* val,
                       __bf16* A, __bf16* Wt, float* Y, float* stats, float* out) {
  cheb_gemm(stream, xin, M, V, Ci, Co, /*Npad=*/Co, /*Nout=*/Co,
            w, /*bias=*/nullptr, idx, val, A, Wt, Y);
  stats_kernel<<<Co, 256, 0, stream>>>(Y, stats, M, Co);
  int n = M * Co;
  bn_relu_kernel<<<cdiv(n, 256), 256, 0, stream>>>(Y, stats, g, be, out, M, Co);
}

extern "C" void kernel_launch(void* const* d_in, const int* in_sizes, int n_in,
                              void* d_out, int out_size, void* d_ws, size_t ws_size,
                              hipStream_t stream) {
  (void)in_sizes; (void)n_in; (void)out_size; (void)ws_size;

  // -------- inputs (setup_inputs dict order) --------
  const float* x = (const float*)d_in[0];
  // blocks: w,b,g,be at indices 1+4*blk ... ; uconv13: w@41, b@42
  const float* W[11];  const float* G[10]; const float* BE[10];
  for (int blk = 0; blk < 10; ++blk) {
    W[blk]  = (const float*)d_in[1 + 4 * blk + 0];
    G[blk]  = (const float*)d_in[1 + 4 * blk + 2];
    BE[blk] = (const float*)d_in[1 + 4 * blk + 3];
  }
  W[10] = (const float*)d_in[41];
  const float* b13 = (const float*)d_in[42];
  const int*   i0 = (const int*)d_in[43]; const float* v0 = (const float*)d_in[44];
  const int*   i1 = (const int*)d_in[45]; const float* v1 = (const float*)d_in[46];
  const int*   i2 = (const int*)d_in[47]; const float* v2 = (const float*)d_in[48];

  const int B = 8;
  const int V0 = 3072, V1 = 768, V2 = 192;
  const int M0 = B * V0, M1 = B * V1, M2 = B * V2;

  // -------- workspace carving --------
  char* ws = (char*)d_ws;
  size_t off = 0;
  auto take = [&](size_t bytes) -> void* {
    void* p = ws + off;
    off += (bytes + 255) & ~(size_t)255;
    return p;
  };
  __bf16* A    = (__bf16*)take((size_t)M0 * 1024 * 2);   // max Kd=1024 @ M0 (50 MB)
  float*  Y    = (float*) take((size_t)M0 * 128 * 4);    // max M*Co f32 (12.6 MB)
  float*  actA = (float*) take((size_t)M0 * 256 * 4);    // 25 MB
  float*  actB = (float*) take((size_t)M0 * 256 * 4);    // 25 MB
  float*  e1   = (float*) take((size_t)M0 * 128 * 4);    // skip level 0
  float*  e2   = (float*) take((size_t)M1 * 256 * 4);    // skip level 1
  int*    idx1 = (int*)   take((size_t)B * V1 * 128 * 4);
  int*    idx2 = (int*)   take((size_t)B * V2 * 256 * 4);
  __bf16* Wt   = (__bf16*)take((size_t)256 * 2048 * 2);  // max packed W^T
  float*  st   = (float*) take((size_t)2 * 512 * 4);     // BN stats

  // -------- encoder --------
  conv_block(stream, x,    M0, V0,   8,  64, W[0], G[0], BE[0], i0, v0, A, Wt, Y, st, actA); // conv11
  conv_block(stream, actA, M0, V0,  64, 128, W[1], G[1], BE[1], i0, v0, A, Wt, Y, st, e1);   // conv13
  pool_kernel<<<cdiv(B * V1 * 128, 256), 256, 0, stream>>>(e1, actA, idx1, B, V0, 128);

  conv_block(stream, actA, M1, V1, 128, 192, W[2], G[2], BE[2], i1, v1, A, Wt, Y, st, actB); // conv21
  conv_block(stream, actB, M1, V1, 192, 256, W[3], G[3], BE[3], i1, v1, A, Wt, Y, st, e2);   // conv23
  pool_kernel<<<cdiv(B * V2 * 256, 256), 256, 0, stream>>>(e2, actA, idx2, B, V1, 256);

  conv_block(stream, actA, M2, V2, 256, 512, W[4], G[4], BE[4], i2, v2, A, Wt, Y, st, actB); // conv31
  conv_block(stream, actB, M2, V2, 512, 256, W[5], G[5], BE[5], i2, v2, A, Wt, Y, st, actA); // conv33

  // -------- decoder --------
  unpool_concat_kernel<<<cdiv(B * V1 * 512, 256), 256, 0, stream>>>(
      actA, idx2, e2, actB, B, V2, 256, 256);
  conv_block(stream, actB, M1, V1, 512, 256, W[6], G[6], BE[6], i1, v1, A, Wt, Y, st, actA); // uconv21
  conv_block(stream, actA, M1, V1, 256, 128, W[7], G[7], BE[7], i1, v1, A, Wt, Y, st, actB); // uconv22

  unpool_concat_kernel<<<cdiv(B * V0 * 256, 256), 256, 0, stream>>>(
      actB, idx1, e1, actA, B, V1, 128, 128);
  conv_block(stream, actA, M0, V0, 256, 128, W[8], G[8], BE[8], i0, v0, A, Wt, Y, st, actB); // uconv11
  conv_block(stream, actB, M0, V0, 128,  64, W[9], G[9], BE[9], i0, v0, A, Wt, Y, st, actA); // uconv12

  // final cheb_conv (no BN/ReLU), Co=2 padded to 16, bias added in store
  cheb_gemm(stream, actA, M0, V0, 64, /*Co=*/2, /*Npad=*/16, /*Nout=*/2,
            W[10], b13, i0, v0, A, Wt, (float*)d_out);
}
